// NeuralODELayer_55551107006715
// MI455X (gfx1250) — compile-verified, matrix operand order checked by reading, not run on hardware
//
#include <hip/hip_runtime.h>
#include <hip/hip_bf16.h>

// ---------------------------------------------------------------------------
// Types for CDNA5 WMMA (gfx1250, wave32)
// ---------------------------------------------------------------------------
typedef __bf16 v16bf __attribute__((ext_vector_type(16)));
typedef __bf16 v8bf  __attribute__((ext_vector_type(8)));
typedef float  v8f   __attribute__((ext_vector_type(8)));
typedef int    v4i   __attribute__((vector_size(16)));

typedef __hip_bfloat16 bhalf;

union FragU {
  v16bf v;
  v8bf  h[2];
};

struct __align__(8) bhalf4 { bhalf x, y, z, w; };

static constexpr int kHID   = 1024;
static constexpr int kBATCH = 8192;
static constexpr int kSTEPS = 10;

// ---------------------------------------------------------------------------
// CDNA5 async global->LDS copy (ASYNCcnt-tracked, no VGPR staging).
// Probe-confirmed signature: (v4i AS1* src, v4i AS3* dst, imm offset, imm cpol)
// Guarded: falls back to a plain LDS copy if the builtin is absent.
// ---------------------------------------------------------------------------
#if defined(__has_builtin)
#  if __has_builtin(__builtin_amdgcn_global_load_async_to_lds_b128)
#    define GEMM_ASYNC_LDS 1
#  endif
#  if __has_builtin(__builtin_amdgcn_s_wait_asynccnt)
#    define HAVE_WAIT_ASYNC_BUILTIN 1
#  endif
#endif

typedef __attribute__((address_space(1))) v4i* gptr_v4i;
typedef __attribute__((address_space(3))) v4i* lptr_v4i;

__device__ __forceinline__ void async_copy_b128(const bhalf* __restrict__ g,
                                                bhalf* __restrict__ l) {
#if GEMM_ASYNC_LDS
  bhalf* gnc = const_cast<bhalf*>(g);
  __builtin_amdgcn_global_load_async_to_lds_b128(
      (gptr_v4i)gnc, (lptr_v4i)l, /*imm offset*/ 0, /*cpol*/ 0);
#else
  *reinterpret_cast<v8bf*>(l) = *reinterpret_cast<const v8bf*>(g);
#endif
}

__device__ __forceinline__ void wait_async_lds() {
#if GEMM_ASYNC_LDS
#  if HAVE_WAIT_ASYNC_BUILTIN
  __builtin_amdgcn_s_wait_asynccnt(0);
#  else
  asm volatile("s_wait_asynccnt 0x0" ::: "memory");
#  endif
#endif
}

// ---------------------------------------------------------------------------
// GEMM: C[M,N] = act(A[M,K] @ B[K,N] + bias (+ t*trow))
// A: bf16 row-major MxK.  Bt: bf16 N x K (pre-transposed weights).
// Block tile 256x128, 256 threads = 8 waves (4 M x 2 N), wave tile 64x64
// (4x4 f32 16x16 accumulators), K-step 64, double-buffered LDS with
// async-to-LDS copy overlapping WMMA; one barrier per K-tile.
// 16 ds_load_b128 per 16 WMMAs per wave per 32-K step.
// ---------------------------------------------------------------------------
template <bool TANH, bool OUT_BF16, bool HAS_T>
__launch_bounds__(256)
__global__ void gemm_bias_act(const bhalf* __restrict__ A,
                              const bhalf* __restrict__ Bt,
                              const float* __restrict__ bias,
                              const float* __restrict__ trow,
                              float t,
                              void* __restrict__ Cout,
                              int M, int N, int K) {
  // 72 = 64 + 8 pad elements; 144B row stride (16B aligned, conflict-free)
  __shared__ __align__(16) bhalf As[2][256][72];   // 2 x 36 KB
  __shared__ __align__(16) bhalf Bs[2][128][72];   // 2 x 18 KB

  const int tid   = threadIdx.x;
  const int lane  = tid & 31;
  const int wave  = tid >> 5;
  const int waveM = wave >> 1;          // 0..3 -> M offset waveM*64
  const int waveN = wave & 1;           // 0..1 -> N offset waveN*64
  const int mBase = blockIdx.y * 256;
  const int nBase = blockIdx.x * 128;
  const int lm    = lane & 15;
  const int lh    = (lane >> 4) << 3;   // 0 (lanes 0-15) or 8 (lanes 16-31)
  (void)M;

  v8f acc[4][4];
#pragma unroll
  for (int i = 0; i < 4; ++i)
#pragma unroll
    for (int j = 0; j < 4; ++j)
#pragma unroll
      for (int r = 0; r < 8; ++r) acc[i][j][r] = 0.0f;

  // A tile: 256x64 -> 2048 16B chunks -> 8/thread. B tile: 128x64 -> 4/thread.
  auto copy_tile = [&](int k0, int buf) {
#pragma unroll
    for (int i = 0; i < 8; ++i) {
      const int chunk = tid + i * 256;
      const int row   = chunk >> 3;
      const int col   = (chunk & 7) << 3;
      async_copy_b128(A + (size_t)(mBase + row) * K + k0 + col, &As[buf][row][col]);
    }
#pragma unroll
    for (int i = 0; i < 4; ++i) {
      const int chunk = tid + i * 256;
      const int row   = chunk >> 3;
      const int col   = (chunk & 7) << 3;
      async_copy_b128(Bt + (size_t)(nBase + row) * K + k0 + col, &Bs[buf][row][col]);
    }
  };

  const int nTiles = K >> 6;            // K/64
  copy_tile(0, 0);

  for (int it = 0; it < nTiles; ++it) {
    const int buf = it & 1;
    wait_async_lds();                   // own tile-it async writes landed
    __syncthreads();                    // everyone's tile-it visible; prev buf free

    if (it + 1 < nTiles) copy_tile((it + 1) << 6, buf ^ 1);  // overlaps WMMAs

#pragma unroll
    for (int kk = 0; kk < 64; kk += 32) {
      // lanes 0-15: K{kk..kk+7, kk+16..kk+23}; lanes 16-31: +8
      const int kb = kk + lh;

      FragU a[4];
#pragma unroll
      for (int i = 0; i < 4; ++i) {
        const bhalf* rp = &As[buf][waveM * 64 + i * 16 + lm][kb];
        a[i].h[0] = *reinterpret_cast<const v8bf*>(rp);
        a[i].h[1] = *reinterpret_cast<const v8bf*>(rp + 16);
      }
#pragma unroll
      for (int j = 0; j < 4; ++j) {
        const bhalf* rp = &Bs[buf][waveN * 64 + j * 16 + lm][kb];
        FragU b;
        b.h[0] = *reinterpret_cast<const v8bf*>(rp);
        b.h[1] = *reinterpret_cast<const v8bf*>(rp + 16);
#pragma unroll
        for (int i = 0; i < 4; ++i)
          acc[i][j] = __builtin_amdgcn_wmma_f32_16x16x32_bf16(
              false, a[i].v, false, b.v, (short)0, acc[i][j], false, false);
      }
    }
  }

  // --- epilogue: bias (+ t*trow), optional tanh, store f32 or bf16 ---
#pragma unroll
  for (int j = 0; j < 4; ++j) {
    const int col = nBase + waveN * 64 + j * 16 + lm;
    float bv = bias[col];
    if (HAS_T) bv += t * trow[col];
#pragma unroll
    for (int i = 0; i < 4; ++i) {
      const int row0 = mBase + waveM * 64 + i * 16 + lh;
#pragma unroll
      for (int r = 0; r < 8; ++r) {
        float v = acc[i][j][r] + bv;
        if (TANH) v = tanhf(v);
        const size_t off = (size_t)(row0 + r) * N + col;
        if (OUT_BF16) reinterpret_cast<bhalf*>(Cout)[off] = __float2bfloat16(v);
        else          reinterpret_cast<float*>(Cout)[off] = v;
      }
    }
  }
}

// ---------------------------------------------------------------------------
// Weight prep: fp32 KxN -> bf16 NxK (transpose) via LDS tile.
// ---------------------------------------------------------------------------
__global__ void transpose_to_bf16(const float* __restrict__ W,
                                  bhalf* __restrict__ Wt, int K, int N) {
  __shared__ float tile[32][33];
  const int n0 = blockIdx.x * 32;
  const int k0 = blockIdx.y * 32;
  const int tx = threadIdx.x;
#pragma unroll
  for (int i = threadIdx.y; i < 32; i += 8)
    tile[i][tx] = W[(size_t)(k0 + i) * N + n0 + tx];
  __syncthreads();
#pragma unroll
  for (int i = threadIdx.y; i < 32; i += 8)
    Wt[(size_t)(n0 + i) * K + k0 + tx] = __float2bfloat16(tile[tx][i]);
}

__global__ void copy_trow(const float* __restrict__ W1, float* __restrict__ trow) {
  const int i = blockIdx.x * blockDim.x + threadIdx.x;
  if (i < kHID) trow[i] = W1[(size_t)kHID * kHID + i];
}

// ---------------------------------------------------------------------------
// H = h_in (fp32 running state in d_out) and Xin = bf16(h_in)
// ---------------------------------------------------------------------------
__global__ void init_state(const float4* __restrict__ hin,
                           float4* __restrict__ H,
                           bhalf* __restrict__ Xin, int n4) {
  const int i = blockIdx.x * blockDim.x + threadIdx.x;
  if (i >= n4) return;
  const float4 v = hin[i];
  H[i] = v;
  bhalf4 o;
  o.x = __float2bfloat16(v.x); o.y = __float2bfloat16(v.y);
  o.z = __float2bfloat16(v.z); o.w = __float2bfloat16(v.w);
  *reinterpret_cast<bhalf4*>(Xin + 4 * (size_t)i) = o;
}

// ---------------------------------------------------------------------------
// RK4 combine:
//   acc_new = accA*ACC + accB*K        (accA==0 -> ignore stale ACC)
//   FINAL:   H += dt6*acc_new; Xin = bf16(H)
//   else:    ACC = acc_new;    Xin = bf16(H + aScale*K)
// ---------------------------------------------------------------------------
template <bool FINAL>
__global__ void rk4_combine(const float4* __restrict__ Kv,
                            float4* __restrict__ H,
                            float4* __restrict__ ACC,
                            bhalf* __restrict__ Xin,
                            float accA, float accB, float aScale, float dt6, int n4) {
  const int i = blockIdx.x * blockDim.x + threadIdx.x;
  if (i >= n4) return;
  const float4 k = Kv[i];
  float4 h = H[i];
  float4 acc;
  if (accA == 0.0f) {
    acc.x = accB * k.x; acc.y = accB * k.y; acc.z = accB * k.z; acc.w = accB * k.w;
  } else {
    const float4 a = ACC[i];
    acc.x = accA * a.x + accB * k.x; acc.y = accA * a.y + accB * k.y;
    acc.z = accA * a.z + accB * k.z; acc.w = accA * a.w + accB * k.w;
  }
  float4 x;
  if (FINAL) {
    h.x += dt6 * acc.x; h.y += dt6 * acc.y; h.z += dt6 * acc.z; h.w += dt6 * acc.w;
    H[i] = h;
    x = h;
  } else {
    ACC[i] = acc;
    x.x = h.x + aScale * k.x; x.y = h.y + aScale * k.y;
    x.z = h.z + aScale * k.z; x.w = h.w + aScale * k.w;
  }
  bhalf4 o;
  o.x = __float2bfloat16(x.x); o.y = __float2bfloat16(x.y);
  o.z = __float2bfloat16(x.z); o.w = __float2bfloat16(x.w);
  *reinterpret_cast<bhalf4*>(Xin + 4 * (size_t)i) = o;
}

// ---------------------------------------------------------------------------
// Launcher
// ---------------------------------------------------------------------------
extern "C" void kernel_launch(void* const* d_in, const int* in_sizes, int n_in,
                              void* d_out, int out_size, void* d_ws, size_t ws_size,
                              hipStream_t stream) {
  (void)in_sizes; (void)n_in; (void)out_size; (void)ws_size;

  const float* h_in = (const float*)d_in[0];
  const float* W1   = (const float*)d_in[1];   // (1025, 1024)
  const float* b1   = (const float*)d_in[2];
  const float* W2   = (const float*)d_in[3];
  const float* b2   = (const float*)d_in[4];
  const float* W3   = (const float*)d_in[5];
  const float* b3   = (const float*)d_in[6];
  float* H = (float*)d_out;                    // running fp32 state

  const float dt = (1.0f - 0.0f) / kSTEPS;

  // ---- workspace carve-out ----
  size_t off = 0;
  auto alloc = [&](size_t bytes) -> void* {
    void* r = (char*)d_ws + off;
    off += (bytes + 255) & ~(size_t)255;
    return r;
  };
  bhalf* W1t  = (bhalf*)alloc((size_t)kHID * kHID * sizeof(bhalf));
  bhalf* W2t  = (bhalf*)alloc((size_t)kHID * kHID * sizeof(bhalf));
  bhalf* W3t  = (bhalf*)alloc((size_t)kHID * kHID * sizeof(bhalf));
  float* trow = (float*)alloc((size_t)kHID * sizeof(float));
  float* ACC  = (float*)alloc((size_t)kBATCH * kHID * sizeof(float));
  float* Kbuf = (float*)alloc((size_t)kBATCH * kHID * sizeof(float));
  bhalf* XA   = (bhalf*)alloc((size_t)kBATCH * kHID * sizeof(bhalf)); // dyn input / X2
  bhalf* XB   = (bhalf*)alloc((size_t)kBATCH * kHID * sizeof(bhalf)); // X1

  // ---- prep: transpose+convert weights (fp32 KxN -> bf16 NxK), t-row ----
  {
    dim3 tb(32, 8), tg(kHID / 32, kHID / 32);
    transpose_to_bf16<<<tg, tb, 0, stream>>>(W1, W1t, kHID, kHID); // first 1024 rows
    transpose_to_bf16<<<tg, tb, 0, stream>>>(W2, W2t, kHID, kHID);
    transpose_to_bf16<<<tg, tb, 0, stream>>>(W3, W3t, kHID, kHID);
    copy_trow<<<(kHID + 255) / 256, 256, 0, stream>>>(W1, trow);
  }

  const int n4 = kBATCH * kHID / 4;
  const int eBlocks = n4 / 256;
  init_state<<<eBlocks, 256, 0, stream>>>((const float4*)h_in, (float4*)H, XA, n4);

  dim3 ggrid(kHID / 128, kBATCH / 256);  // (8, 32) = 256 workgroups
  dim3 gblock(256);

  auto run_dyn = [&](float t) {
    // x1 = tanh(Xin @ W1 + b1 + t*W1[last])   -> XB (bf16)
    gemm_bias_act<true, true, true><<<ggrid, gblock, 0, stream>>>(
        XA, W1t, b1, trow, t, (void*)XB, kBATCH, kHID, kHID);
    // x2 = tanh(x1 @ W2 + b2)                 -> XA (bf16, reuse)
    gemm_bias_act<true, true, false><<<ggrid, gblock, 0, stream>>>(
        XB, W2t, b2, nullptr, 0.0f, (void*)XA, kBATCH, kHID, kHID);
    // k  = x2 @ W3 + b3                       -> Kbuf (fp32)
    gemm_bias_act<false, false, false><<<ggrid, gblock, 0, stream>>>(
        XA, W3t, b3, nullptr, 0.0f, (void*)Kbuf, kBATCH, kHID, kHID);
  };

  for (int s = 0; s < kSTEPS; ++s) {
    const float t0 = dt * (float)s;
    // k1
    run_dyn(t0);
    rk4_combine<false><<<eBlocks, 256, 0, stream>>>(
        (const float4*)Kbuf, (float4*)H, (float4*)ACC, XA,
        0.0f, 1.0f, 0.5f * dt, 0.0f, n4);
    // k2
    run_dyn(t0 + 0.5f * dt);
    rk4_combine<false><<<eBlocks, 256, 0, stream>>>(
        (const float4*)Kbuf, (float4*)H, (float4*)ACC, XA,
        1.0f, 2.0f, 0.5f * dt, 0.0f, n4);
    // k3
    run_dyn(t0 + 0.5f * dt);
    rk4_combine<false><<<eBlocks, 256, 0, stream>>>(
        (const float4*)Kbuf, (float4*)H, (float4*)ACC, XA,
        1.0f, 2.0f, dt, 0.0f, n4);
    // k4 + finalize step
    run_dyn(t0 + dt);
    rk4_combine<true><<<eBlocks, 256, 0, stream>>>(
        (const float4*)Kbuf, (float4*)H, (float4*)ACC, XA,
        1.0f, 1.0f, 0.0f, dt / 6.0f, n4);
  }
}
